// EfficientTransformerBlock3D_28398323761338
// MI455X (gfx1250) — matrix operand order repaired and checked by reading, not compiled, not run on hardware
//
#include <hip/hip_runtime.h>
#include <hip/hip_bf16.h>

typedef _Float16 f16;
typedef __attribute__((ext_vector_type(16))) _Float16 v16h;
typedef __attribute__((ext_vector_type(8)))  _Float16 v8h;
typedef __attribute__((ext_vector_type(8)))  float    v8f;

#define OUT_F32   0
#define OUT_F32T  1
#define OUT_F16   2
#define OUT_F16T  3

#define WMMA_F16(A, B, C) \
    __builtin_amdgcn_wmma_f32_16x16x32_f16(false, (A), false, (B), (short)0, (C), false, false)

// ---------------------------------------------------------------------------
// Generic f16 WMMA GEMM:  Out[m,n] = sum_k A[m,k]*B[n,k]  (+bias[n]) (+res[m,n])
// A: [M,K] f16 row-major, B: [Nout,K] f16 row-major.
// M%128==0, Nout%64==0, K%64==0, K>=64.
// Wave computes 32(M) x 64(N) (2 A-frags x 4 B-frags -> 8 WMMA / K-step);
// block = 4 waves -> 128x64 tile. K-loop is a copy-free ping-pong pipeline
// with a peeled tail: fragment loads for one buffer always issue underneath
// the 8 WMMAs of the other buffer.
// ---------------------------------------------------------------------------
__global__ __launch_bounds__(128)
void gemm_f16_wmma(const f16* __restrict__ A, long long strideA,
                   const f16* __restrict__ Bm, long long strideB,
                   int M, int Nout, int K,
                   const float* __restrict__ bias,
                   const float* __restrict__ residual, long long strideRes,
                   float* __restrict__ outF32, f16* __restrict__ outF16,
                   long long strideOut, int outMode)
{
    const int batch = blockIdx.z;
    const f16* Ab = A  + (long long)batch * strideA;
    const f16* Bb = Bm + (long long)batch * strideB;
    const float* res = residual ? residual + (long long)batch * strideRes : nullptr;
    float* oF = outF32 ? outF32 + (long long)batch * strideOut : nullptr;
    f16*   oH = outF16 ? outF16 + (long long)batch * strideOut : nullptr;

    const int wave = threadIdx.x >> 5;
    const int lane = threadIdx.x & 31;
    const int mBase = blockIdx.x * 128 + wave * 32;
    const int nBase = blockIdx.y * 64;
    const int l15   = lane & 15;          // A row-in-tile / B col-in-tile / C-N
    const int hk    = (lane >> 4) * 16;   // A half-K offset (contiguous K, matches B)
    const int kh16  = (lane >> 4) * 16;   // B half-K offset

    // Per-lane base pointers (already include the half-K offsets).
    const f16* aRow0 = Ab + (long long)(mBase + l15) * K + hk;
    const f16* aRow1 = aRow0 + 16LL * K;
    const f16* bR0 = Bb + (long long)(nBase +  0 + l15) * K + kh16;
    const f16* bR1 = Bb + (long long)(nBase + 16 + l15) * K + kh16;
    const f16* bR2 = Bb + (long long)(nBase + 32 + l15) * K + kh16;
    const f16* bR3 = Bb + (long long)(nBase + 48 + l15) * K + kh16;

    auto ldA = [](const f16* p) {
        v8h lo = *(const v8h*)(p);        // K = hk + 0..7
        v8h hi = *(const v8h*)(p + 8);    // K = hk + 8..15
        v16h r;
#pragma unroll
        for (int j = 0; j < 8; ++j) { r[j] = lo[j]; r[8 + j] = hi[j]; }
        return r;
    };

    struct Frags { v16h a0, a1, b0, b1, b2, b3; };
    auto ldFrags = [&](int k) {
        Frags f;
        f.a0 = ldA(aRow0 + k);
        f.a1 = ldA(aRow1 + k);
        f.b0 = *(const v16h*)(bR0 + k);
        f.b1 = *(const v16h*)(bR1 + k);
        f.b2 = *(const v16h*)(bR2 + k);
        f.b3 = *(const v16h*)(bR3 + k);
        return f;
    };

    v8f acc[8] = {};
    auto doWMMA = [&](const Frags& f) {
        acc[0] = WMMA_F16(f.a0, f.b0, acc[0]);
        acc[1] = WMMA_F16(f.a0, f.b1, acc[1]);
        acc[2] = WMMA_F16(f.a0, f.b2, acc[2]);
        acc[3] = WMMA_F16(f.a0, f.b3, acc[3]);
        acc[4] = WMMA_F16(f.a1, f.b0, acc[4]);
        acc[5] = WMMA_F16(f.a1, f.b1, acc[5]);
        acc[6] = WMMA_F16(f.a1, f.b2, acc[6]);
        acc[7] = WMMA_F16(f.a1, f.b3, acc[7]);
    };

    // Ping-pong pipeline, peeled tail (K >= 64, K % 64 == 0).
    Frags fA = ldFrags(0);
    int k0 = 0;
    for (; k0 + 64 < K; k0 += 64) {
        // Speculative prefetch 4 K-steps ahead (invalid translations dropped).
        __builtin_prefetch(aRow0 + k0 + 128, 0, 1);
        __builtin_prefetch(bR0 + k0 + 128, 0, 1);
        Frags fB = ldFrags(k0 + 32);
        doWMMA(fA);                 // overlaps fB's loads
        fA = ldFrags(k0 + 64);      // redefined after last use -> no reg copies
        doWMMA(fB);                 // overlaps fA's loads
    }
    {
        Frags fB = ldFrags(k0 + 32);
        doWMMA(fA);
        doWMMA(fB);
    }

    // Epilogue. C/D layout: m = r + (lane>>4)*8, n = lane&15.
    const int mOff = (lane >> 4) * 8;
#pragma unroll
    for (int mi = 0; mi < 2; ++mi) {
#pragma unroll
        for (int nSub = 0; nSub < 4; ++nSub) {
            const v8f av = acc[mi * 4 + nSub];
            const int n = nBase + nSub * 16 + l15;
            const float bv = bias ? bias[n] : 0.0f;
#pragma unroll
            for (int r = 0; r < 8; ++r) {
                const int m = mBase + mi * 16 + mOff + r;
                float v = av[r] + bv;
                if (res) v += res[(long long)m * Nout + n];
                if (outMode == OUT_F32)       oF[(long long)m * Nout + n] = v;
                else if (outMode == OUT_F32T) oF[(long long)n * M + m]    = v;
                else if (outMode == OUT_F16)  oH[(long long)m * Nout + n] = (f16)v;
                else                          oH[(long long)n * M + m]    = (f16)v;
            }
        }
    }
}

// ---------------------------------------------------------------------------
// LayerNorm over C=256 per token; one wave32 per token, 8 channels per lane.
// ---------------------------------------------------------------------------
__global__ __launch_bounds__(256)
void ln_kernel(const float* __restrict__ x, const float* __restrict__ g,
               const float* __restrict__ b, f16* __restrict__ out, int nTokens)
{
    const int wave = threadIdx.x >> 5, lane = threadIdx.x & 31;
    const int tok = blockIdx.x * 8 + wave;
    if (tok >= nTokens) return;
    const float* xr = x + (long long)tok * 256;
    float vals[8];
    float s = 0.f;
#pragma unroll
    for (int i = 0; i < 8; ++i) { vals[i] = xr[lane + 32 * i]; s += vals[i]; }
#pragma unroll
    for (int off = 16; off > 0; off >>= 1) s += __shfl_xor(s, off, 32);
    const float mean = s * (1.0f / 256.0f);
    float vs = 0.f;
#pragma unroll
    for (int i = 0; i < 8; ++i) { float d = vals[i] - mean; vs += d * d; }
#pragma unroll
    for (int off = 16; off > 0; off >>= 1) vs += __shfl_xor(vs, off, 32);
    const float inv = rsqrtf(vs * (1.0f / 256.0f) + 1e-5f);
    f16* o = out + (long long)tok * 256;
#pragma unroll
    for (int i = 0; i < 8; ++i) {
        const int c = lane + 32 * i;
        o[c] = (f16)((vals[i] - mean) * inv * g[c] + b[c]);
    }
}

// ---------------------------------------------------------------------------
// Softmax along a long row (spatial softmax of keys): one block per row.
// ---------------------------------------------------------------------------
__global__ __launch_bounds__(256)
void softmax_row_kernel(const float* __restrict__ in, f16* __restrict__ out, int L)
{
    __shared__ float sred[8];
    __shared__ float sbc;
    const long long row = blockIdx.x;
    const float* r = in + row * (long long)L;
    f16* o = out + row * (long long)L;
    const int lane = threadIdx.x & 31, wave = threadIdx.x >> 5;

    float mx = -3.0e38f;
    for (int i = threadIdx.x; i < L; i += 256) mx = fmaxf(mx, r[i]);
#pragma unroll
    for (int off = 16; off > 0; off >>= 1) mx = fmaxf(mx, __shfl_xor(mx, off, 32));
    if (lane == 0) sred[wave] = mx;
    __syncthreads();
    if (threadIdx.x == 0) {
        float m = sred[0];
        for (int i = 1; i < 8; ++i) m = fmaxf(m, sred[i]);
        sbc = m;
    }
    __syncthreads();
    mx = sbc;
    __syncthreads();

    float s = 0.f;
    for (int i = threadIdx.x; i < L; i += 256) s += __expf(r[i] - mx);
#pragma unroll
    for (int off = 16; off > 0; off >>= 1) s += __shfl_xor(s, off, 32);
    if (lane == 0) sred[wave] = s;
    __syncthreads();
    if (threadIdx.x == 0) {
        float t = 0.f;
        for (int i = 0; i < 8; ++i) t += sred[i];
        sbc = 1.0f / t;
    }
    __syncthreads();
    const float inv = sbc;
    for (int i = threadIdx.x; i < L; i += 256) o[i] = (f16)(__expf(r[i] - mx) * inv);
}

// ---------------------------------------------------------------------------
// Softmax over 256 channels per token (queries): one wave32 per token.
// ---------------------------------------------------------------------------
__global__ __launch_bounds__(256)
void softmax_token_kernel(const float* __restrict__ in, f16* __restrict__ out, int nTokens)
{
    const int wave = threadIdx.x >> 5, lane = threadIdx.x & 31;
    const int tok = blockIdx.x * 8 + wave;
    if (tok >= nTokens) return;
    const float* r = in + (long long)tok * 256;
    float v[8];
    float mx = -3.0e38f;
#pragma unroll
    for (int i = 0; i < 8; ++i) { v[i] = r[lane + 32 * i]; mx = fmaxf(mx, v[i]); }
#pragma unroll
    for (int off = 16; off > 0; off >>= 1) mx = fmaxf(mx, __shfl_xor(mx, off, 32));
    float s = 0.f;
#pragma unroll
    for (int i = 0; i < 8; ++i) { v[i] = __expf(v[i] - mx); s += v[i]; }
#pragma unroll
    for (int off = 16; off > 0; off >>= 1) s += __shfl_xor(s, off, 32);
    const float inv = 1.0f / s;
    f16* o = out + (long long)tok * 256;
#pragma unroll
    for (int i = 0; i < 8; ++i) o[lane + 32 * i] = (f16)(v[i] * inv);
}

// ---------------------------------------------------------------------------
// Depthwise 3x3x3 conv (SAME) + bias + exact GELU, sliding window along W.
// One block per (b,d,h) row; thread keeps the 9 (kd,kh) neighbor columns in
// registers and slides across W: 9 loads/output instead of 27.
// h1/out: [B,N,hid] f16 with n = (d*32+h)*32+w.
// ---------------------------------------------------------------------------
__global__ __launch_bounds__(256)
void dwconv_gelu_kernel(const f16* __restrict__ h1, const float* __restrict__ w,
                        const float* __restrict__ bias, f16* __restrict__ out)
{
    const int dh = blockIdx.x;           // 0..511 = d*32+h
    const int b  = blockIdx.y;
    const int d  = dh >> 5;
    const int h  = dh & 31;
    const long long bbase = (long long)b * 16384;

    int  rowIdx[9];
    bool rowOk[9];
#pragma unroll
    for (int kd = -1; kd <= 1; ++kd) {
#pragma unroll
        for (int kh = -1; kh <= 1; ++kh) {
            const int j = (kd + 1) * 3 + (kh + 1);
            const int dd = d + kd, hh = h + kh;
            rowOk[j]  = (dd >= 0 && dd < 16 && hh >= 0 && hh < 32);
            rowIdx[j] = (dd << 10) | (hh << 5);
        }
    }
    const int centerRow = (d << 10) | (h << 5);

    for (int ch = threadIdx.x; ch < 1024; ch += 256) {
        const float* wc = w + ch * 27;
        const float bch = bias[ch];
        float c0[9], c1[9], c2[9];
#pragma unroll
        for (int j = 0; j < 9; ++j) {
            c0[j] = 0.0f;
            c1[j] = rowOk[j] ? (float)h1[(bbase + rowIdx[j]) * 1024 + ch] : 0.0f;
        }
        for (int ww = 0; ww < 32; ++ww) {
#pragma unroll
            for (int j = 0; j < 9; ++j)
                c2[j] = (ww + 1 < 32 && rowOk[j])
                      ? (float)h1[(bbase + rowIdx[j] + ww + 1) * 1024 + ch] : 0.0f;
            float acc = bch;
#pragma unroll
            for (int j = 0; j < 9; ++j)
                acc += c0[j] * wc[j * 3 + 0] + c1[j] * wc[j * 3 + 1] + c2[j] * wc[j * 3 + 2];
            const float g = 0.5f * acc * (1.0f + erff(acc * 0.70710678118654752f));
            out[(bbase + centerRow + ww) * 1024 + ch] = (f16)g;
#pragma unroll
            for (int j = 0; j < 9; ++j) { c0[j] = c1[j]; c1[j] = c2[j]; }
        }
    }
}

// ---------------------------------------------------------------------------
__global__ __launch_bounds__(256)
void cvt_f32_f16_kernel(const float* __restrict__ in, f16* __restrict__ out, long long n)
{
    const long long i = (long long)blockIdx.x * 256 + threadIdx.x;
    if (i < n) out[i] = (f16)in[i];
}

// ---------------------------------------------------------------------------
extern "C" void kernel_launch(void* const* d_in, const int* in_sizes, int n_in,
                              void* d_out, int out_size, void* d_ws, size_t ws_size,
                              hipStream_t stream)
{
    (void)in_sizes; (void)n_in; (void)out_size; (void)ws_size;
    const float* x     = (const float*)d_in[0];
    const float* ln1_g = (const float*)d_in[1];
    const float* ln1_b = (const float*)d_in[2];
    const float* Wk    = (const float*)d_in[3];
    const float* bk    = (const float*)d_in[4];
    const float* Wq    = (const float*)d_in[5];
    const float* bq    = (const float*)d_in[6];
    const float* Wv    = (const float*)d_in[7];
    const float* bv    = (const float*)d_in[8];
    const float* Wr    = (const float*)d_in[9];
    const float* br    = (const float*)d_in[10];
    const float* ln2_g = (const float*)d_in[11];
    const float* ln2_b = (const float*)d_in[12];
    const float* fc1W  = (const float*)d_in[13];
    const float* fc1b  = (const float*)d_in[14];
    const float* dwW   = (const float*)d_in[15];
    const float* dwb   = (const float*)d_in[16];
    const float* fc2W  = (const float*)d_in[17];
    const float* fc2b  = (const float*)d_in[18];
    float* out = (float*)d_out;

    constexpr int  B = 2, C = 256, HID = 1024;
    constexpr long long N  = 16384;        // D*H*W = 16*32*32
    constexpr long long NC = N * C;
    constexpr long long NH = N * HID;
    constexpr long long CC = (long long)C * C;

    // ---- workspace carve-up ----
    char* p = (char*)d_ws;
    f16* n1    = (f16*)p; p += B * NC * sizeof(f16);
    f16* n2    = (f16*)p; p += B * NC * sizeof(f16);
    f16* qsm   = (f16*)p; p += B * NC * sizeof(f16);
    f16* ksm   = (f16*)p; p += B * NC * sizeof(f16);
    f16* valT  = (f16*)p; p += B * NC * sizeof(f16);
    f16* h1    = (f16*)p; p += B * NH * sizeof(f16);
    f16* act   = (f16*)p; p += B * NH * sizeof(f16);
    f16* Wk16  = (f16*)p; p += CC * sizeof(f16);
    f16* Wq16  = (f16*)p; p += CC * sizeof(f16);
    f16* Wv16  = (f16*)p; p += CC * sizeof(f16);
    f16* Wr16  = (f16*)p; p += CC * sizeof(f16);
    f16* fc1W16 = (f16*)p; p += (long long)HID * C * sizeof(f16);
    f16* fc2W16 = (f16*)p; p += (long long)C * HID * sizeof(f16);
    f16* ctx16  = (f16*)p; p += B * CC * sizeof(f16);
    f16* Mfold  = (f16*)p; p += B * CC * sizeof(f16);
    float* keysT = (float*)p; p += B * NC * sizeof(float);
    float* qf    = (float*)p; p += B * NC * sizeof(float);
    float* tx    = (float*)p; p += B * NC * sizeof(float);

    float* ctx_out = out + B * NC;   // context lives at the tail of d_out

    // ---- 1. weights -> f16 ----
    cvt_f32_f16_kernel<<<(CC + 255) / 256, 256, 0, stream>>>(Wk, Wk16, CC);
    cvt_f32_f16_kernel<<<(CC + 255) / 256, 256, 0, stream>>>(Wq, Wq16, CC);
    cvt_f32_f16_kernel<<<(CC + 255) / 256, 256, 0, stream>>>(Wv, Wv16, CC);
    cvt_f32_f16_kernel<<<(CC + 255) / 256, 256, 0, stream>>>(Wr, Wr16, CC);
    cvt_f32_f16_kernel<<<((long long)HID * C + 255) / 256, 256, 0, stream>>>(fc1W, fc1W16, (long long)HID * C);
    cvt_f32_f16_kernel<<<((long long)C * HID + 255) / 256, 256, 0, stream>>>(fc2W, fc2W16, (long long)C * HID);

    // ---- 2. LN1 ----
    ln_kernel<<<(B * N + 7) / 8, 256, 0, stream>>>(x, ln1_g, ln1_b, n1, (int)(B * N));

    // ---- 3. QKV GEMMs [N,C]x[C,C] ----
    dim3 gNC(N / 128, C / 64, B), blk(128);
    gemm_f16_wmma<<<gNC, blk, 0, stream>>>(n1, NC, Wk16, 0, (int)N, C, C,
                                           bk, nullptr, 0, keysT, nullptr, NC, OUT_F32T);
    gemm_f16_wmma<<<gNC, blk, 0, stream>>>(n1, NC, Wq16, 0, (int)N, C, C,
                                           bq, nullptr, 0, qf, nullptr, NC, OUT_F32);
    gemm_f16_wmma<<<gNC, blk, 0, stream>>>(n1, NC, Wv16, 0, (int)N, C, C,
                                           bv, nullptr, 0, nullptr, valT, NC, OUT_F16T);

    // ---- 4. softmaxes ----
    softmax_row_kernel<<<B * C, 256, 0, stream>>>(keysT, ksm, (int)N);
    softmax_token_kernel<<<(B * N + 7) / 8, 256, 0, stream>>>(qf, qsm, (int)(B * N));

    // ---- 5. context[k,v] = sum_n ksm[k,n]*valT[v,n]  (K = 16384) -> d_out tail ----
    dim3 gCtx(C / 128, C / 64, B);
    gemm_f16_wmma<<<gCtx, blk, 0, stream>>>(ksm, NC, valT, NC, C, C, (int)N,
                                            nullptr, nullptr, 0, ctx_out, nullptr, CC, OUT_F32);
    cvt_f32_f16_kernel<<<(B * CC + 255) / 256, 256, 0, stream>>>(ctx_out, ctx16, B * CC);

    // ---- 6. fold reprojection: M[c,k] = sum_v Wr[c,v]*ctx[k,v] -> f16 [C,C] ----
    gemm_f16_wmma<<<gCtx, blk, 0, stream>>>(Wr16, 0, ctx16, CC, C, C, C,
                                            nullptr, nullptr, 0, nullptr, Mfold, CC, OUT_F16);

    // ---- 7. attn + residual: tx[n,c] = sum_k qsm[n,k]*M[c,k] + br[c] + x[n,c] ----
    gemm_f16_wmma<<<gNC, blk, 0, stream>>>(qsm, NC, Mfold, CC, (int)N, C, C,
                                           br, x, NC, tx, nullptr, NC, OUT_F32);

    // ---- 8. LN2 ----
    ln_kernel<<<(B * N + 7) / 8, 256, 0, stream>>>(tx, ln2_g, ln2_b, n2, (int)(B * N));

    // ---- 9. fc1: [N,C]x[HID,C] -> h1 f16 [N,HID] ----
    dim3 gFc1(N / 128, HID / 64, B);
    gemm_f16_wmma<<<gFc1, blk, 0, stream>>>(n2, NC, fc1W16, 0, (int)N, HID, C,
                                            fc1b, nullptr, 0, nullptr, h1, NH, OUT_F16);

    // ---- 10. depthwise conv + GELU (sliding window along W) ----
    dim3 gConv(512, B);
    dwconv_gelu_kernel<<<gConv, 256, 0, stream>>>(h1, dwW, dwb, act);

    // ---- 11. fc2 + residual -> mx into d_out ----
    gemm_f16_wmma<<<gNC, blk, 0, stream>>>(act, NH, fc2W16, 0, (int)N, C, HID,
                                           fc2b, tx, NC, out, nullptr, NC, OUT_F32);
}
